// GLR_77120432767409
// MI455X (gfx1250) — matrix operand order repaired and checked by reading, not compiled, not run on hardware
//
#include <hip/hip_runtime.h>
#include <math.h>

typedef __attribute__((ext_vector_type(16))) _Float16 v16h;
typedef __attribute__((ext_vector_type(8)))  _Float16 v8h;
typedef __attribute__((ext_vector_type(8)))  float    v8f;

#define NPX 1296
#define GW  36
#define UMAXV 15.5625f

__constant__ int c_dr8[8] = {-1,-1,-1, 0, 0, 1, 1, 1};
__constant__ int c_dc8[8] = {-1, 0, 1,-1, 1,-1, 0, 1};

// ---------------------------------------------------------------------------
// Weight pre-pack: OIHW f32 -> f16 A-fragments.
// Layout: [tap][mt][k0][lane][h] contiguous, 512 halves per (tap,mt,k0).
// A (16x32 f16) lane map: lanes 0-15 row M=lane,    K = {0..15}
//                         lanes 16-31 row M=lane-16, K = {16..31}
// ---------------------------------------------------------------------------
__global__ void pack_w_k(const float* __restrict__ w, _Float16* __restrict__ dst,
                         int Cout, int Cin, int nMt, int nK, int total)
{
    int idx = blockIdx.x * blockDim.x + threadIdx.x;
    if (idx >= total) return;
    int h    = idx & 15;
    int lane = (idx >> 4) & 31;
    int rest = idx >> 9;                 // (tap*nMt + mt)*nK + k0
    int k0   = rest % nK;
    int mt   = (rest / nK) % nMt;
    int tap  = rest / (nK * nMt);
    int m    = lane & 15;
    int hi   = lane >> 4;
    int mco  = mt * 16 + m;
    int K    = k0 * 32 + hi * 16 + h;
    float v  = (mco < Cout && K < Cin) ? w[((size_t)mco * Cin + K) * 9 + tap] : 0.0f;
    dst[idx] = (_Float16)v;
}

// xf f32 NCHW (16,3,36,36) -> f16 NHWC padded to 32 channels (zeros)
__global__ void cvt_xf_k(const float* __restrict__ xf, _Float16* __restrict__ out, int total)
{
    int idx = blockIdx.x * blockDim.x + threadIdx.x;
    if (idx >= total) return;
    int c = idx & 31;
    int p = (idx >> 5) % NPX;
    int b = idx / (32 * NPX);
    float v = (c < 3) ? xf[((size_t)b * 3 + c) * NPX + p] : 0.0f;
    out[idx] = (_Float16)v;
}

// ---------------------------------------------------------------------------
// Implicit-GEMM 3x3 conv on WMMA, NHWC f16 activations (CinP mult of 32).
// One wave32 computes a 16(Cout) x 16(pos) tile; NK = Cin/32 chunks (template).
// Uniform per-block fast path (no masking) for interior tiles / pad==0;
// boundary tiles use one predicated load block per tap.
// ---------------------------------------------------------------------------
template<int NK>
__global__ __launch_bounds__(32)
void conv3x3_wmma(const _Float16* __restrict__ in, const _Float16* __restrict__ wpk,
                  const float* __restrict__ bias, _Float16* __restrict__ out,
                  int CinP, int Hin, int Win,
                  int Cout, int Hout, int Wout, int CtotP, int coff,
                  int stride, int pad, int do_relu)
{
    const int lane = threadIdx.x;
    const int m    = lane & 15;
    const int hi   = lane >> 4;
    const int b    = blockIdx.z;
    const int mt   = blockIdx.y;
    const int nMt  = gridDim.y;
    const int HWo  = Hout * Wout;
    const int p0   = blockIdx.x * 16;
    const int p    = p0 + m;
    const bool pok = p < HWo;
    const int orow = pok ? (p / Wout) : 0;
    const int ocol = pok ? (p % Wout) : 0;
    const int ir0  = orow * stride - pad;
    const int ic0  = ocol * stride - pad;
    const _Float16* ibase = in + (size_t)b * Hin * Win * CinP + hi * 16;

    // ---- uniform (blockIdx-only) interior test ----
    bool fast;
    if (pad == 0) {
        fast = true;   // clamped positions always index valid memory; dead cols unsaved
    } else {
        const int pL = p0 + 15;
        if (pL < HWo) {
            const int r0 = p0 / Wout, c0 = p0 % Wout;
            const int r1 = pL / Wout, c1 = pL % Wout;
            fast = (r0 == r1) &&
                   (r0 * stride - pad) >= 0 && (r1 * stride - pad + 2) < Hin &&
                   (c0 * stride - pad) >= 0 && (c1 * stride - pad + 2) < Win;
        } else {
            fast = false;
        }
    }

    v8f acc = {};
    if (fast) {
        for (int tap = 0; tap < 9; ++tap) {
            const int rr = ir0 + tap / 3, cc = ic0 + tap % 3;
            const _Float16* prow = ibase + ((size_t)rr * Win + cc) * CinP;
            const _Float16* wrow = wpk + ((((size_t)tap * nMt + mt) * NK) << 9) + (lane << 4);
#pragma unroll
            for (int k0 = 0; k0 < NK; ++k0) {
                v16h afr = *(const v16h*)(wrow + ((size_t)k0 << 9));
                v16h bfr = *(const v16h*)(prow + (k0 << 5));
                acc = __builtin_amdgcn_wmma_f32_16x16x32_f16(false, afr, false, bfr,
                                                             (short)0, acc, false, false);
            }
        }
    } else {
        for (int tap = 0; tap < 9; ++tap) {
            const int rr = ir0 + tap / 3, cc = ic0 + tap % 3;
            const bool inb = pok && rr >= 0 && rr < Hin && cc >= 0 && cc < Win;
            const _Float16* prow = ibase + ((size_t)rr * Win + cc) * CinP;
            const _Float16* wrow = wpk + ((((size_t)tap * nMt + mt) * NK) << 9) + (lane << 4);
            v16h bf[NK];
            const v16h zero = {};
#pragma unroll
            for (int k0 = 0; k0 < NK; ++k0) bf[k0] = zero;
            if (inb) {
#pragma unroll
                for (int k0 = 0; k0 < NK; ++k0)
                    bf[k0] = *(const v16h*)(prow + (k0 << 5));
            }
#pragma unroll
            for (int k0 = 0; k0 < NK; ++k0) {
                v16h afr = *(const v16h*)(wrow + ((size_t)k0 << 9));
                acc = __builtin_amdgcn_wmma_f32_16x16x32_f16(false, afr, false, bf[k0],
                                                             (short)0, acc, false, false);
            }
        }
    }

    if (pok) {
        const int c0 = mt * 16 + hi * 8;
        v8h hv;
#pragma unroll
        for (int v = 0; v < 8; ++v) {
            const int co = c0 + v;
            float val = (co < Cout) ? (acc[v] + bias[co]) : 0.0f;
            if (do_relu) val = fmaxf(val, 0.0f);
            hv[v] = (_Float16)val;
        }
        *(v8h*)(out + ((size_t)b * HWo + p) * CtotP + coff + c0) = hv;
    }
}

// ---------------------------------------------------------------------------
// NHWC f16 elementwise helpers
// ---------------------------------------------------------------------------
__global__ void maxpool2_k(const _Float16* __restrict__ in, _Float16* __restrict__ out,
                           int Hin, int Win, int Hout, int Wout, int C, int total)
{
    int idx = blockIdx.x * blockDim.x + threadIdx.x;
    if (idx >= total) return;
    int c  = idx % C;  int t = idx / C;
    int wo = t % Wout; t /= Wout;
    int ho = t % Hout; int b = t / Hout;
    float mx = -3.0e38f;
#pragma unroll
    for (int dy = 0; dy < 2; ++dy)
#pragma unroll
        for (int dx = 0; dx < 2; ++dx) {
            int r = ho * 2 + dy, cc = wo * 2 + dx;
            if (r < Hin && cc < Win)
                mx = fmaxf(mx, (float)in[(((size_t)b * Hin + r) * Win + cc) * C + c]);
        }
    out[idx] = (_Float16)mx;
}

// upsample x2 (nearest) + reflect-pad 1, NHWC: out is (2H+2, 2W+2)
__global__ void up2refpad_k(const _Float16* __restrict__ in, _Float16* __restrict__ out,
                            int Hin, int Win, int C, int total)
{
    int Ho = 2 * Hin + 2, Wo = 2 * Win + 2;
    int idx = blockIdx.x * blockDim.x + threadIdx.x;
    if (idx >= total) return;
    int ch = idx % C; int t = idx / C;
    int c  = t % Wo;  t /= Wo;
    int r  = t % Ho;  int b = t / Ho;
    int ur = r - 1; if (ur < 0) ur = 1; else if (ur > 2 * Hin - 1) ur = 2 * Hin - 2;
    int uc = c - 1; if (uc < 0) uc = 1; else if (uc > 2 * Win - 1) uc = 2 * Win - 2;
    out[idx] = in[(((size_t)b * Hin + (ur >> 1)) * Win + (uc >> 1)) * C + ch];
}

__global__ void concat_copy_k(const _Float16* __restrict__ src, _Float16* __restrict__ dst,
                              int C, int HW, int Ctot, int coff, int total)
{
    int idx = blockIdx.x * blockDim.x + threadIdx.x;
    if (idx >= total) return;
    int c = idx % C; int t = idx / C;
    int p = t % HW;  int b = t / HW;
    dst[((size_t)b * HW + p) * Ctot + coff + c] = src[idx];
}

// E: NHWC f16 (pad32) -> NCHW f32
__global__ void cvt_e_k(const _Float16* __restrict__ eh, float* __restrict__ eb, int total)
{
    int idx = blockIdx.x * blockDim.x + threadIdx.x;
    if (idx >= total) return;
    int p = idx % NPX;
    int c = (idx / NPX) % 3;
    int b = idx / (3 * NPX);
    eb[idx] = (float)eh[((size_t)b * NPX + p) * 32 + c];
}

// Y = x + cnny_out : NHWC f16 conv output + NCHW f32 xf -> NCHW f32
__global__ void cvt_y_k(const _Float16* __restrict__ yh, const float* __restrict__ xf,
                        float* __restrict__ yb, int total)
{
    int idx = blockIdx.x * blockDim.x + threadIdx.x;
    if (idx >= total) return;
    int p = idx % NPX;
    int c = (idx / NPX) % 3;
    int b = idx / (3 * NPX);
    yb[idx] = (float)yh[((size_t)b * NPX + p) * 32 + c] + xf[idx];
}

// cnnu head: pooled NHWC f16 (B,3,3,32) -> fc1(32x288, NCHW-flat) -> fc2 -> min(UMAX)
__global__ __launch_bounds__(32)
void cnnu_fc_k(const _Float16* __restrict__ pooled,
               const float* __restrict__ fc1w, const float* __restrict__ fc1b,
               const float* __restrict__ fc2w, const float* __restrict__ fc2b,
               float* __restrict__ u)
{
    int b = blockIdx.x, k = threadIdx.x;
    float s = fc1b[k];
    for (int j = 0; j < 288; ++j) {
        int c = j / 9, pix = j % 9;                       // NCHW flatten order
        float xv = (float)pooled[((size_t)b * 9 + pix) * 32 + c];
        s += xv * fc1w[k * 288 + j];
    }
    float t = s * fc2w[k];
#pragma unroll
    for (int o = 16; o > 0; o >>= 1) t += __shfl_down(t, o, 32);
    if (k == 0) u[b] = fminf(t + fc2b[0], UMAXV);
}

// Edge weights of the 8-connected grid: W8[b][k][i] = exp(-||E_i - E_j||^2 / 2)
__global__ void edge_w_k(const float* __restrict__ E, float* __restrict__ W8,
                         float* __restrict__ deg, int total)
{
    int idx = blockIdx.x * blockDim.x + threadIdx.x;
    if (idx >= total) return;
    int i = idx % NPX, b = idx / NPX;
    const float* Eb = E + (size_t)b * 3 * NPX;
    float e0 = Eb[i], e1 = Eb[NPX + i], e2 = Eb[2 * NPX + i];
    int r = i / GW, c = i % GW;
    float ds = 0.0f;
#pragma unroll
    for (int k = 0; k < 8; ++k) {
        int nr = r + c_dr8[k], nc = c + c_dc8[k];
        float w = 0.0f;
        if (nr >= 0 && nr < GW && nc >= 0 && nc < GW) {
            int j = nr * GW + nc;
            float a0 = e0 - Eb[j], a1 = e1 - Eb[NPX + j], a2 = e2 - Eb[2 * NPX + j];
            w = expf(-0.5f * (a0 * a0 + a1 * a1 + a2 * a2));
        }
        W8[((size_t)b * 8 + k) * NPX + i] = w;
        ds += w;
    }
    deg[idx] = ds;
}

// ---------------------------------------------------------------------------
// CG solve of (I + u*L) x = y per (batch, rhs) system, fully LDS-resident.
// ---------------------------------------------------------------------------
__device__ __forceinline__ float blk_reduce(float v, float* red)
{
    int tid = threadIdx.x;
    red[tid] = v;
    __syncthreads();
    for (int s = 128; s > 0; s >>= 1) {
        if (tid < s) red[tid] += red[tid + s];
        __syncthreads();
    }
    float r = red[0];
    __syncthreads();
    return r;
}

__global__ __launch_bounds__(256)
void cg_solve_k(const float* __restrict__ Y, const float* __restrict__ W8,
                const float* __restrict__ deg, const float* __restrict__ u,
                float* __restrict__ out)
{
    __shared__ float xs[NPX], rs[NPX], ps[NPX], aps[NPX], red[256];
    const int b = blockIdx.x / 3, ch = blockIdx.x % 3;
    const int tid = threadIdx.x;
    const float uu = u[b];
    const float* Wb = W8 + (size_t)b * 8 * NPX;
    const float* db = deg + (size_t)b * NPX;
    const float* yb = Y + (size_t)b * 3 * NPX;

    for (int i = tid; i < NPX; i += 256) {
        float y = yb[3 * i + ch];                 // reference reshape semantics
        xs[i] = 0.0f; rs[i] = y; ps[i] = y;
    }
    __syncthreads();

    float part = 0.0f;
    for (int i = tid; i < NPX; i += 256) part += rs[i] * rs[i];
    float rr = blk_reduce(part, red);
    const float tol = fmaxf(rr * 1e-14f, 1e-30f);

    for (int it = 0; it < 300; ++it) {
        for (int i = tid; i < NPX; i += 256) {
            const int r = i / GW, c = i % GW;
            const float pv = ps[i];
            float s = db[i] * pv;
#pragma unroll
            for (int k = 0; k < 8; ++k) {
                const int nr = r + c_dr8[k], nc = c + c_dc8[k];
                if (nr >= 0 && nr < GW && nc >= 0 && nc < GW)
                    s -= Wb[k * NPX + i] * ps[nr * GW + nc];
            }
            aps[i] = pv + uu * s;
        }
        float pp = 0.0f;
        for (int i = tid; i < NPX; i += 256) pp += ps[i] * aps[i];
        float pap = blk_reduce(pp, red);
        if (!(pap > 0.0f)) break;
        const float alpha = rr / pap;
        float rn = 0.0f;
        for (int i = tid; i < NPX; i += 256) {
            xs[i] += alpha * ps[i];
            const float rv = rs[i] - alpha * aps[i];
            rs[i] = rv;
            rn += rv * rv;
        }
        float rrn = blk_reduce(rn, red);
        if (rrn < tol) break;
        const float beta = rrn / rr;
        rr = rrn;
        for (int i = tid; i < NPX; i += 256) ps[i] = rs[i] + beta * ps[i];
        __syncthreads();
    }
    for (int i = tid; i < NPX; i += 256)
        out[(size_t)b * 3 * NPX + 3 * i + ch] = xs[i];
}

// ---------------------------------------------------------------------------
extern "C" void kernel_launch(void* const* d_in, const int* in_sizes, int n_in,
                              void* d_out, int out_size, void* d_ws, size_t ws_size,
                              hipStream_t stream)
{
    (void)in_sizes; (void)n_in; (void)out_size; (void)ws_size;
    const float* xf = (const float*)d_in[0];
    enum { F1A = 0, F1B, F2A, F2B, F3A, F3B, FD1, F4A, F4B, FD2, F5A, F5B, F5C,
           Y1, Y2, Y3, Y4, U1, U2, U3, U4, NSPEC };
    struct Spec { int cout, cin; };
    const Spec sp[NSPEC] = {{32,3},{32,32},{64,32},{64,64},{64,64},{128,64},{64,128},
                            {64,128},{64,64},{32,64},{32,64},{32,32},{3,32},
                            {32,3},{32,32},{32,32},{3,32},
                            {32,3},{32,32},{32,32},{32,32}};
    auto Wraw = [&](int s) { return (const float*)d_in[1 + 2 * s]; };
    auto Bp   = [&](int s) { return (const float*)d_in[2 + 2 * s]; };
    const float* fc1w = (const float*)d_in[43];
    const float* fc1b = (const float*)d_in[44];
    const float* fc2w = (const float*)d_in[45];
    const float* fc2b = (const float*)d_in[46];

    // byte bump allocator, 64B aligned (keeps v16h loads 32B aligned)
    char* base = (char*)d_ws;
    size_t off = 0;
    auto allocB = [&](size_t bytes) { bytes = (bytes + 63) & ~(size_t)63;
                                      char* p = base + off; off += bytes; return (void*)p; };
    auto allocH = [&](size_t n) { return (_Float16*)allocB(n * 2); };
    auto allocF = [&](size_t n) { return (float*)allocB(n * 4); };

    // f16 NHWC activation buffers (channels padded to 32)
    _Float16* XH   = allocH(663552);   // xf NHWC pad32
    _Float16* T1h  = allocH(663552);   // 32ch @36^2
    _Float16* T2h  = allocH(1478656);  // f2a out / l3u(20^2x128) / l4u(38^2x64)
    _Float16* T3h  = allocH(331776);   // 64ch @18^2
    _Float16* T4h  = allocH(82944);    // 64ch @9^2
    _Float16* T5h  = allocH(165888);   // 128ch @9^2 / 32ch @18^2
    _Float16* L1h  = allocH(663552);   // l1
    _Float16* L2h  = allocH(331776);   // l2 / l4b
    _Float16* CAT1 = allocH(663552);   // 128ch @18^2 / 32ch @36^2
    _Float16* CAT2 = allocH(1327104);  // 64ch @36^2
    _Float16* EHh  = allocH(663552);   // E NHWC pad32
    _Float16* YHh  = allocH(663552);   // cnny conv out NHWC pad32
    _Float16* P1h  = allocH(12800);    // 32ch @5^2
    _Float16* P2h  = allocH(12800);
    _Float16* P3h  = allocH(4608);     // 32ch @3^2
    // f32 buffers
    float* EB   = allocF(62208);
    float* YB   = allocF(62208);
    float* Wbuf = allocF(165888);
    float* DEG  = allocF(20736);
    float* UU   = allocF(16);
    // packed f16 weights
    _Float16* wpk[NSPEC];
    for (int s = 0; s < NSPEC; ++s) {
        int nMt = (sp[s].cout + 15) / 16, nK = (sp[s].cin + 31) / 32;
        int tot = 9 * nMt * nK * 512;
        wpk[s] = allocH(tot);
        pack_w_k<<<(tot + 255) / 256, 256, 0, stream>>>(Wraw(s), wpk[s],
                                                        sp[s].cout, sp[s].cin, nMt, nK, tot);
    }
    cvt_xf_k<<<(663552 + 255) / 256, 256, 0, stream>>>(xf, XH, 663552);

    auto conv = [&](const _Float16* in, int s, _Float16* out,
                    int CinP, int Hin, int Win, int Hout, int Wout,
                    int CtotP, int coff, int stride, int pad, int relu) {
        dim3 g((Hout * Wout + 15) / 16, (sp[s].cout + 15) / 16, 16);
        int nK = (sp[s].cin + 31) / 32;
        if (nK == 1)
            conv3x3_wmma<1><<<g, dim3(32), 0, stream>>>(in, wpk[s], Bp(s), out,
                CinP, Hin, Win, sp[s].cout, Hout, Wout, CtotP, coff, stride, pad, relu);
        else if (nK == 2)
            conv3x3_wmma<2><<<g, dim3(32), 0, stream>>>(in, wpk[s], Bp(s), out,
                CinP, Hin, Win, sp[s].cout, Hout, Wout, CtotP, coff, stride, pad, relu);
        else
            conv3x3_wmma<4><<<g, dim3(32), 0, stream>>>(in, wpk[s], Bp(s), out,
                CinP, Hin, Win, sp[s].cout, Hout, Wout, CtotP, coff, stride, pad, relu);
    };
    auto pool = [&](const _Float16* in, _Float16* out, int Hi, int Wi, int Ho, int Wo, int C) {
        int tot = 16 * Ho * Wo * C;
        maxpool2_k<<<(tot + 255) / 256, 256, 0, stream>>>(in, out, Hi, Wi, Ho, Wo, C, tot);
    };
    auto upref = [&](const _Float16* in, _Float16* out, int Hi, int Wi, int C) {
        int tot = 16 * (2 * Hi + 2) * (2 * Wi + 2) * C;
        up2refpad_k<<<(tot + 255) / 256, 256, 0, stream>>>(in, out, Hi, Wi, C, tot);
    };
    auto ccopy = [&](const _Float16* src, _Float16* dst, int C, int HW, int Ctot, int coff) {
        int tot = 16 * C * HW;
        concat_copy_k<<<(tot + 255) / 256, 256, 0, stream>>>(src, dst, C, HW, Ctot, coff, tot);
    };

    // ---------------- cnnf ----------------
    conv(XH,   F1A, T1h, 32, 36, 36, 36, 36, 32, 0, 1, 1, 1);
    conv(T1h,  F1B, L1h, 32, 36, 36, 36, 36, 32, 0, 1, 1, 1);
    conv(L1h,  F2A, T2h, 32, 36, 36, 36, 36, 64, 0, 1, 1, 1);
    pool(T2h, T3h, 36, 36, 18, 18, 64);
    conv(T3h,  F2B, L2h, 64, 18, 18, 18, 18, 64, 0, 1, 1, 1);
    conv(L2h,  F3A, T3h, 64, 18, 18, 18, 18, 64, 0, 1, 1, 1);
    pool(T3h, T4h, 18, 18, 9, 9, 64);
    conv(T4h,  F3B, T5h, 64, 9, 9, 9, 9, 128, 0, 1, 1, 1);
    upref(T5h, T2h, 9, 9, 128);                                    // (20,20,128)
    conv(T2h,  FD1, CAT1, 128, 20, 20, 18, 18, 128, 0, 1, 0, 0);   // pad=0
    ccopy(L2h, CAT1, 64, 324, 128, 64);
    conv(CAT1, F4A, T3h, 128, 18, 18, 18, 18, 64, 0, 1, 1, 1);
    conv(T3h,  F4B, L2h, 64, 18, 18, 18, 18, 64, 0, 1, 1, 1);
    upref(L2h, T2h, 18, 18, 64);                                   // (38,38,64)
    conv(T2h,  FD2, CAT2, 64, 38, 38, 36, 36, 64, 0, 1, 0, 0);
    ccopy(L1h, CAT2, 32, 1296, 64, 32);
    conv(CAT2, F5A, T1h, 64, 36, 36, 36, 36, 32, 0, 1, 1, 1);
    conv(T1h,  F5B, CAT1, 32, 36, 36, 36, 36, 32, 0, 1, 1, 1);
    conv(CAT1, F5C, EHh, 32, 36, 36, 36, 36, 32, 0, 1, 1, 0);      // E (3 of 32 ch)

    // ---------------- cnny ----------------
    conv(XH,   Y1, T1h, 32, 36, 36, 36, 36, 32, 0, 1, 1, 1);
    conv(T1h,  Y2, CAT1, 32, 36, 36, 36, 36, 32, 0, 1, 1, 1);
    conv(CAT1, Y3, T1h, 32, 36, 36, 36, 36, 32, 0, 1, 1, 1);
    conv(T1h,  Y4, YHh, 32, 36, 36, 36, 36, 32, 0, 1, 1, 0);
    cvt_y_k<<<(62208 + 255) / 256, 256, 0, stream>>>(YHh, xf, YB, 62208);

    // ---------------- cnnu ----------------
    conv(XH,  U1, T5h, 32, 36, 36, 18, 18, 32, 0, 2, 1, 1);        // stride 2
    conv(T5h, U2, T3h, 32, 18, 18, 18, 18, 32, 0, 1, 1, 1);
    pool(T3h, T4h, 18, 18, 9, 9, 32);
    conv(T4h, U3, T3h, 32, 9, 9, 9, 9, 32, 0, 1, 1, 1);
    pool(T3h, P1h, 9, 9, 5, 5, 32);                                // ceil
    conv(P1h, U4, P2h, 32, 5, 5, 5, 5, 32, 0, 1, 1, 1);
    pool(P2h, P3h, 5, 5, 3, 3, 32);                                // ceil
    cnnu_fc_k<<<16, 32, 0, stream>>>(P3h, fc1w, fc1b, fc2w, fc2b, UU);

    // ---------------- graph weights + CG solve ----------------
    cvt_e_k<<<(62208 + 255) / 256, 256, 0, stream>>>(EHh, EB, 62208);
    edge_w_k<<<(16 * NPX + 255) / 256, 256, 0, stream>>>(EB, Wbuf, DEG, 16 * NPX);
    cg_solve_k<<<48, 256, 0, stream>>>(YB, Wbuf, DEG, UU, (float*)d_out);
}
